// FusionModule_63780264345704
// MI455X (gfx1250) — compile-verified, matrix operand order checked by reading, not run on hardware
//
#include <hip/hip_runtime.h>

typedef __attribute__((ext_vector_type(16))) __bf16 v16bf;
typedef __attribute__((ext_vector_type(4)))  __bf16 v4bf;
typedef __attribute__((ext_vector_type(8)))  float  v8f;

#define T_STEPS 5
#define BATCH   8
#define CCH     64
#define HDIM    128
#define WDIM    128
#define HP      130                                // padded
#define WP      130                                // padded
#define IMGS    (T_STEPS * BATCH)                  // 40
#define PIX     (HDIM * WDIM)                      // 16384
#define PPIX    (HP * WP)                          // 16900
#define IMG_ELEMS   ((long)PIX * CCH)              // 1,048,576
#define PIMG_ELEMS  ((long)PPIX * CCH)             // 1,081,600
#define STEP_ELEMS  ((long)BATCH * IMG_ELEMS)      // 8,388,608
#define TOTAL_ELEMS ((long)IMGS * IMG_ELEMS)       // 41,943,040
#define PTOTAL_ELEMS ((long)IMGS * PIMG_ELEMS)     // 43,264,000

#define V_TH    0.1f
#define V_CLAMP 2.0f

// ---------------------------------------------------------------------------
// 0) Zero-fill a padded bf16 buffer (halo must be zero): 16 B per thread.
// ---------------------------------------------------------------------------
__global__ __launch_bounds__(256) void fill_zero_kernel(float4* __restrict__ p) {
  int i = blockIdx.x * 256 + threadIdx.x;           // over PTOTAL_ELEMS/8
  p[i] = make_float4(0.f, 0.f, 0.f, 0.f);
}

// ---------------------------------------------------------------------------
// 1) x fp32 NCHW -> bf16 padded pixel-major [img, 1+y, 1+x, c]
// ---------------------------------------------------------------------------
__global__ __launch_bounds__(256) void cvt_x_kernel(const float* __restrict__ x,
                                                    __bf16* __restrict__ xp) {
  long i = (long)blockIdx.x * 256 + threadIdx.x;    // NCHW linear index
  int xc  = (int)(i & (WDIM - 1));
  int y   = (int)((i >> 7)  & (HDIM - 1));
  int c   = (int)((i >> 14) & (CCH - 1));
  int img = (int)(i >> 20);
  long o = ((long)img * PPIX + (long)(y + 1) * WP + (xc + 1)) * CCH + c;
  xp[o] = (__bf16)x[i];
}

// ---------------------------------------------------------------------------
// 2) Repack weights [Cout,Cin,3,3] fp32 -> bf16 WMMA B-fragments.
//    wf[((f*4 + cot)*32 + lane)*16 + j], f = (ky*3+kx)*2 + chalf.
//    16-bit operand layout: lane<16 holds K 0..7 & 16..23; lane>=16: 8..15 & 24..31.
// ---------------------------------------------------------------------------
__global__ __launch_bounds__(256) void build_wfrag_kernel(const float* __restrict__ w,
                                                          __bf16* __restrict__ wf) {
  int i = blockIdx.x * 256 + threadIdx.x;           // 0 .. 36863
  int j    = i & 15;
  int lane = (i >> 4) & 31;
  int cot  = (i >> 9) & 3;
  int f    = i >> 11;                               // 0..17
  int ch   = f & 1;
  int pos  = f >> 1;                                // 0..8
  int ky = pos / 3, kx = pos % 3;
  int co = cot * 16 + (lane & 15);
  int kb = (lane < 16) ? 0 : 8;
  int koff = (j < 8) ? (kb + j) : (kb + 8 + j);     // j>=8 -> kb+16+(j-8)
  int ci = ch * 32 + koff;
  wf[i] = (__bf16)w[((co * CCH + ci) * 3 + ky) * 3 + kx];
}

// ---------------------------------------------------------------------------
// 3) 3x3 conv (pad 1) as implicit GEMM. Block = 128 thr = 4 waves.
//    Each wave: its own 16-pixel tile x ALL 64 couts (4 f32 accumulators).
//    A fragment loaded once per K-chunk -> 4 WMMAs; B read from LDS stage.
//    72 v_wmma per wave; no branches (padded input).
// ---------------------------------------------------------------------------
__global__ __launch_bounds__(128) void conv3x3_wmma_kernel(
    const __bf16* __restrict__ in,    // padded [img, yp, xp, c] bf16
    const __bf16* __restrict__ wf,    // 36864 bf16 B-fragments
    const float*  __restrict__ bias,  // [64]
    float*        __restrict__ out)   // unpadded [img, y, x, c] fp32
{
  __shared__ __bf16 lwf[36864];                     // 73,728 B

  const int tid  = threadIdx.x;
  const int lane = tid & 31;
  const int wave = tid >> 5;

  // Cooperative stage of the whole fragment table into LDS (b128 copies).
  {
    const float4* src = (const float4*)wf;
    float4* dst = (float4*)lwf;
#pragma unroll
    for (int k = 0; k < 36; ++k)                    // 4608 float4s / 128 thr
      dst[tid + k * 128] = src[tid + k * 128];
  }
  __syncthreads();

  const int bidx = blockIdx.x;
  const int xh   = bidx & 1;                        // 2 half-rows of 64 px
  const int y    = (bidx >> 1) & 127;
  const int img  = bidx >> 8;                       // 0..39
  const int x0   = xh * 64 + wave * 16;             // this wave's pixel tile

  const int m  = lane & 15;                         // A row (pixel) / B col
  const int kb = (lane < 16) ? 0 : 8;               // per-lane K base

  union V32 { v16bf v; float4 f4[2]; };

  v8f acc0, acc1, acc2, acc3;
  {
    const float b0 = bias[m], b1 = bias[16 + m], b2 = bias[32 + m], b3 = bias[48 + m];
#pragma unroll
    for (int r = 0; r < 8; ++r) { acc0[r] = b0; acc1[r] = b1; acc2[r] = b2; acc3[r] = b3; }
  }

  // Per-lane A base: padded row y, col x0+m, channel kb. All tap offsets are
  // compile-time immediates from here (fit 24-bit IOFFSET).
  const __bf16* pbase = in + ((long)img * PPIX + (long)y * WP + (x0 + m)) * CCH + kb;

#pragma unroll
  for (int pos = 0; pos < 9; ++pos) {
    const int ky = pos / 3, kx = pos % 3;
    const int aoff = (ky * WP + kx) * CCH;          // immediate element offset
#pragma unroll
    for (int ch = 0; ch < 2; ++ch) {
      V32 av;
      av.f4[0] = *(const float4*)(pbase + aoff + ch * 32);
      av.f4[1] = *(const float4*)(pbase + aoff + ch * 32 + 16);
      const int f2 = pos * 2 + ch;
#pragma unroll
      for (int cot = 0; cot < 4; ++cot) {
        V32 bv;
        const __bf16* q = lwf + (((f2 * 4 + cot) * 32 + lane) << 4);
        bv.f4[0] = *(const float4*)(q);
        bv.f4[1] = *(const float4*)(q + 8);
        v8f& acc = (cot == 0) ? acc0 : (cot == 1) ? acc1 : (cot == 2) ? acc2 : acc3;
        acc = __builtin_amdgcn_wmma_f32_16x16x32_bf16(
            false, av.v, false, bv.v, (short)0, acc, false, false);
      }
    }
  }

  // C/D layout: VGPR r -> pixel M = r + 8*(lane>=16); col = lane&15 within tile.
  const int mrow = (lane < 16) ? 0 : 8;
  float* op = out + (long)img * IMG_ELEMS + (long)(y * WDIM + x0) * CCH + m;
#pragma unroll
  for (int r = 0; r < 8; ++r) {
    float* orow = op + (long)(mrow + r) * CCH;
    orow[0]  = acc0[r];
    orow[16] = acc1[r];
    orow[32] = acc2[r];
    orow[48] = acc3[r];
  }
}

// ---------------------------------------------------------------------------
// 4) LIF over T (hard reset, clamp). Mid: bf16 spikes into PADDED buffer.
//    4 channels per thread (float4 loads / v4bf stores).
// ---------------------------------------------------------------------------
__global__ __launch_bounds__(256) void lif_mid_kernel(const float* __restrict__ h,
                                                      const float* __restrict__ tau,
                                                      __bf16* __restrict__ spp) {
  const long i = (long)blockIdx.x * 256 + threadIdx.x;   // [0, STEP_ELEMS/4)
  const long e = i << 2;                                  // NHWC element index
  const int c = (int)(e & 63);
  const int x = (int)((e >> 6)  & 127);
  const int y = (int)((e >> 13) & 127);
  const int b = (int)(e >> 20);
  const float inv_tau = 1.0f / tau[0];
  float v0 = 0.f, v1 = 0.f, v2 = 0.f, v3 = 0.f;
#pragma unroll
  for (int t = 0; t < T_STEPS; ++t) {
    const float4 xv = *(const float4*)(h + e + (long)t * STEP_ELEMS);
    v0 += inv_tau * (xv.x - v0);
    v1 += inv_tau * (xv.y - v1);
    v2 += inv_tau * (xv.z - v2);
    v3 += inv_tau * (xv.w - v3);
    const float s0 = (v0 >= V_TH) ? 1.f : 0.f;
    const float s1 = (v1 >= V_TH) ? 1.f : 0.f;
    const float s2 = (v2 >= V_TH) ? 1.f : 0.f;
    const float s3 = (v3 >= V_TH) ? 1.f : 0.f;
    v0 = fminf(fmaxf(v0 * (1.f - s0), -V_CLAMP), V_CLAMP);
    v1 = fminf(fmaxf(v1 * (1.f - s1), -V_CLAMP), V_CLAMP);
    v2 = fminf(fmaxf(v2 * (1.f - s2), -V_CLAMP), V_CLAMP);
    v3 = fminf(fmaxf(v3 * (1.f - s3), -V_CLAMP), V_CLAMP);
    v4bf sv; sv[0] = (__bf16)s0; sv[1] = (__bf16)s1; sv[2] = (__bf16)s2; sv[3] = (__bf16)s3;
    const long img = (long)t * BATCH + b;
    *(v4bf*)(spp + (img * PPIX + (long)(y + 1) * WP + (x + 1)) * CCH + c) = sv;
  }
}

// Final LIF: fp32 spikes scattered to NCHW [T,B,C,H,W] in d_out.
__global__ __launch_bounds__(256) void lif_out_kernel(const float* __restrict__ h,
                                                      const float* __restrict__ tau,
                                                      float* __restrict__ outp) {
  const long i = (long)blockIdx.x * 256 + threadIdx.x;
  const long e = i << 2;
  const int c = (int)(e & 63);
  const int x = (int)((e >> 6)  & 127);
  const int y = (int)((e >> 13) & 127);
  const int b = (int)(e >> 20);
  const float inv_tau = 1.0f / tau[0];
  float v0 = 0.f, v1 = 0.f, v2 = 0.f, v3 = 0.f;
#pragma unroll
  for (int t = 0; t < T_STEPS; ++t) {
    const float4 xv = *(const float4*)(h + e + (long)t * STEP_ELEMS);
    v0 += inv_tau * (xv.x - v0);
    v1 += inv_tau * (xv.y - v1);
    v2 += inv_tau * (xv.z - v2);
    v3 += inv_tau * (xv.w - v3);
    const float s0 = (v0 >= V_TH) ? 1.f : 0.f;
    const float s1 = (v1 >= V_TH) ? 1.f : 0.f;
    const float s2 = (v2 >= V_TH) ? 1.f : 0.f;
    const float s3 = (v3 >= V_TH) ? 1.f : 0.f;
    v0 = fminf(fmaxf(v0 * (1.f - s0), -V_CLAMP), V_CLAMP);
    v1 = fminf(fmaxf(v1 * (1.f - s1), -V_CLAMP), V_CLAMP);
    v2 = fminf(fmaxf(v2 * (1.f - s2), -V_CLAMP), V_CLAMP);
    v3 = fminf(fmaxf(v3 * (1.f - s3), -V_CLAMP), V_CLAMP);
    float* o = outp + (((long)(t * BATCH + b) * CCH + c) * HDIM + y) * WDIM + x;
    o[0L * PIX] = s0; o[1L * PIX] = s1; o[2L * PIX] = s2; o[3L * PIX] = s3;
  }
}

// ---------------------------------------------------------------------------
// ws layout (bytes, 256-aligned):
//   xpad  bf16 @ 0           : 86,528,000
//   sppad bf16 @ 86,528,000  : 86,528,000
//   h     f32  @ 173,056,000 : 167,772,160 (reused by both convs)
//   wf1   bf16 @ 340,828,160 : 73,728
//   wf2   bf16 @ 340,901,888 : 73,728     (total ~325 MiB)
// ---------------------------------------------------------------------------
extern "C" void kernel_launch(void* const* d_in, const int* in_sizes, int n_in,
                              void* d_out, int out_size, void* d_ws, size_t ws_size,
                              hipStream_t stream) {
  const float* x    = (const float*)d_in[0];
  const float* w1   = (const float*)d_in[1];
  const float* b1   = (const float*)d_in[2];
  const float* tau1 = (const float*)d_in[3];
  const float* w2   = (const float*)d_in[4];
  const float* b2   = (const float*)d_in[5];
  const float* tau2 = (const float*)d_in[6];
  float* out = (float*)d_out;

  char* ws = (char*)d_ws;
  __bf16* xpad  = (__bf16*)(ws);
  __bf16* sppad = (__bf16*)(ws + 86528000L);
  float*  hbuf  = (float*) (ws + 173056000L);
  __bf16* wf1   = (__bf16*)(ws + 340828160L);
  __bf16* wf2   = (__bf16*)(ws + 340901888L);

  const int fill_blocks = (int)(PTOTAL_ELEMS / 8 / 256);  // 21,125
  const int cvt_blocks  = (int)(TOTAL_ELEMS / 256);       // 163,840
  const int conv_blocks = IMGS * HDIM * 2;                // 10,240
  const int lif_blocks  = (int)(STEP_ELEMS / 4 / 256);    // 8,192

  fill_zero_kernel<<<fill_blocks, 256, 0, stream>>>((float4*)xpad);
  fill_zero_kernel<<<fill_blocks, 256, 0, stream>>>((float4*)sppad);
  cvt_x_kernel<<<cvt_blocks, 256, 0, stream>>>(x, xpad);
  build_wfrag_kernel<<<144, 256, 0, stream>>>(w1, wf1);
  build_wfrag_kernel<<<144, 256, 0, stream>>>(w2, wf2);

  conv3x3_wmma_kernel<<<conv_blocks, 128, 0, stream>>>(xpad, wf1, b1, hbuf);
  lif_mid_kernel<<<lif_blocks, 256, 0, stream>>>(hbuf, tau1, sppad);
  conv3x3_wmma_kernel<<<conv_blocks, 128, 0, stream>>>(sppad, wf2, b2, hbuf);
  lif_out_kernel<<<lif_blocks, 256, 0, stream>>>(hbuf, tau2, out);
}
